// ResidualMambaBlock_13374528159783
// MI455X (gfx1250) — compile-verified, matrix operand order checked by reading, not run on hardware
//
#include <hip/hip_runtime.h>
#include <hip/hip_bf16.h>

typedef __bf16 bf16_t;
typedef __attribute__((ext_vector_type(16))) __bf16 v16bf;
typedef __attribute__((ext_vector_type(8)))  __bf16 v8bf;
typedef __attribute__((ext_vector_type(8)))  float  v8f;

#define LTOK 16384

__device__ __forceinline__ bf16_t f2bf(float x) {
  unsigned u = __builtin_bit_cast(unsigned, x);
  unsigned r = u + 0x7FFFu + ((u >> 16) & 1u);
  unsigned short s = (unsigned short)(r >> 16);
  return __builtin_bit_cast(bf16_t, s);
}
__device__ __forceinline__ float bf2f(bf16_t b) {
  unsigned short s = __builtin_bit_cast(unsigned short, b);
  unsigned u = ((unsigned)s) << 16;
  return __builtin_bit_cast(float, u);
}

// ---------------------------------------------------------------------------
// Generic bf16 WMMA GEMM: C[M x Npad] (f32) = A[M x K] (bf16, row major, lda)
//                                           @ Bt^T + bias
// Bt is the weight stored TRANSPOSED + padded: Bt[n][k], ldb == Kpad (mult 32).
// One wave computes a 16 x (16*NT) tile: one A fragment feeds NT WMMAs.
// blockDim = 128 (4 waves). grid.z = batch with explicit element strides.
// ---------------------------------------------------------------------------
template <int NT>
__global__ void gemm_bf16_wmma(const bf16_t* __restrict__ A, int lda,
                               const bf16_t* __restrict__ Bt, int ldb,
                               const float* __restrict__ bias, int nBias,
                               float* __restrict__ C, int ldc,
                               int M, int Npad, int Kpad,
                               long sA, long sB, long sC)
{
  int wave = threadIdx.x >> 5;
  int lane = threadIdx.x & 31;
  int tilesM = M >> 4;
  int tilesN = Npad / (16 * NT);
  int tile = blockIdx.x * 4 + wave;
  if (tile >= tilesM * tilesN) return;          // uniform per wave: EXEC stays all-1s
  int tm = tile % tilesM;                       // consecutive waves share the B tile
  int tn = tile / tilesM;
  int half = lane >> 4;
  int lr   = lane & 15;

  const bf16_t* Ab = A  + (long)blockIdx.z * sA;
  const bf16_t* Bb = Bt + (long)blockIdx.z * sB;
  float*        Cb = C  + (long)blockIdx.z * sC;

  const bf16_t* ap = Ab + (long)(tm * 16 + lr) * lda + half * 8;
  const bf16_t* bp[NT];
#pragma unroll
  for (int t = 0; t < NT; ++t)
    bp[t] = Bb + (long)((tn * NT + t) * 16 + lr) * ldb + half * 8;

  union U { v16bf v; v8bf h[2]; };
  v8f zero = {0.f,0.f,0.f,0.f,0.f,0.f,0.f,0.f};
  v8f acc[NT];
#pragma unroll
  for (int t = 0; t < NT; ++t) acc[t] = zero;

  for (int kk = 0; kk < Kpad; kk += 32) {
    U a;
    a.h[0] = *(const v8bf*)(ap + kk);        // lanes 0-15: K=kk..kk+7 ; 16-31: K=kk+8..15
    a.h[1] = *(const v8bf*)(ap + kk + 16);   // lanes 0-15: K=kk+16..23; 16-31: K=kk+24..31
#pragma unroll
    for (int t = 0; t < NT; ++t) {
      U b;
      b.h[0] = *(const v8bf*)(bp[t] + kk);
      b.h[1] = *(const v8bf*)(bp[t] + kk + 16);
      acc[t] = __builtin_amdgcn_wmma_f32_16x16x32_bf16(false, a.v, false, b.v,
                                                       (short)0, acc[t], false, false);
    }
  }
#pragma unroll
  for (int t = 0; t < NT; ++t) {
    int n = (tn * NT + t) * 16 + lr;
    float bv = (n < nBias) ? bias[n] : 0.f;
    float* cp = Cb + (long)(tm * 16 + half * 8) * ldc + n;
#pragma unroll
    for (int j = 0; j < 8; ++j) cp[(long)j * ldc] = acc[t][j] + bv;  // row m = tm*16+8*half+j
  }
}

// ---------------------------------------------------------------------------
// Weight prep: w (K x N, f32) -> wt (Npad x Kpad, bf16, transposed, zero-padded)
// ---------------------------------------------------------------------------
__global__ void prep_weight(const float* __restrict__ w, bf16_t* __restrict__ wt,
                            int K, int N, int Kpad, int Npad)
{
  int idx = blockIdx.x * blockDim.x + threadIdx.x;
  if (idx >= Kpad * Npad) return;
  int n = idx / Kpad, k = idx % Kpad;
  float v = (n < N && k < K) ? w[(long)k * N + n] : 0.f;
  wt[idx] = f2bf(v);
}

// ---------------------------------------------------------------------------
// 3D convs. x: (2,16,16,32,32). Branch conv -> tokens (L x 32) + leaky relu.
// ---------------------------------------------------------------------------
__global__ void conv3d_branch(const float* __restrict__ x, const float* __restrict__ w,
                              const float* __restrict__ bias, float* __restrict__ tok,
                              int b, int ksz)
{
  int idx = blockIdx.x * blockDim.x + threadIdx.x;
  if (idx >= 32 * LTOK) return;
  int l = idx & (LTOK - 1);
  int c = idx >> 14;
  int dz = l >> 10, hz = (l >> 5) & 31, wz = l & 31;
  int pad = ksz >> 1;
  float acc = bias[c];
  for (int ci = 0; ci < 16; ++ci) {
    const float* xp = x + (((long)b * 16 + ci) << 14);
    const float* wp = w + (long)(c * 16 + ci) * ksz * ksz * ksz;
    for (int kd = 0; kd < ksz; ++kd) {
      int d2 = dz + kd - pad; if ((unsigned)d2 >= 16u) continue;
      for (int kh = 0; kh < ksz; ++kh) {
        int h2 = hz + kh - pad; if ((unsigned)h2 >= 32u) continue;
        for (int kw = 0; kw < ksz; ++kw) {
          int w2 = wz + kw - pad; if ((unsigned)w2 >= 32u) continue;
          acc += xp[(d2 << 10) + (h2 << 5) + w2] * wp[(kd * ksz + kh) * ksz + kw];
        }
      }
    }
  }
  tok[(long)l * 32 + c] = acc > 0.f ? acc : 0.01f * acc;
}

__global__ void conv3d_shortcut(const float* __restrict__ x, const float* __restrict__ w,
                                const float* __restrict__ bias, float* __restrict__ out)
{
  int idx = blockIdx.x * blockDim.x + threadIdx.x;
  if (idx >= 2 * 32 * LTOK) return;
  int l  = idx & (LTOK - 1);
  int bc = idx >> 14;
  int c  = bc & 31;
  int b  = bc >> 5;
  float acc = bias[c];
  for (int ci = 0; ci < 16; ++ci)
    acc += x[(((long)b * 16 + ci) << 14) + l] * w[c * 16 + ci];
  out[idx] = acc;                                   // initializes d_out
}

__global__ void accum_out(const float* __restrict__ t, float* __restrict__ outb, int L)
{
  int idx = blockIdx.x * blockDim.x + threadIdx.x;
  if (idx >= 32 * L) return;
  int c = idx >> 14, l = idx & (L - 1);
  outb[idx] += t[(long)l * 32 + c];
}

// ---------------------------------------------------------------------------
// f32 -> bf16 column-slice convert with K padding and optional gelu(tanh).
// ---------------------------------------------------------------------------
__global__ void convert_slice(const float* __restrict__ src, int sld, int scol, int ncol,
                              bf16_t* __restrict__ dst, int dld, int rows, int act)
{
  int idx = blockIdx.x * blockDim.x + threadIdx.x;
  if (idx >= rows * dld) return;
  int c = idx % dld;
  int r = idx / dld;
  float v = 0.f;
  if (c < ncol) {
    v = src[(long)r * sld + scol + c];
    if (act == 1) {  // gelu (tanh approx)
      float t = 0.7978845608f * v * (1.f + 0.044715f * v * v);
      v = 0.5f * v * (1.f + tanhf(t));
    }
  }
  dst[idx] = f2bf(v);
}

// ---------------------------------------------------------------------------
// LayerNorm over 64 features -> bf16. One wave32 per row, 2 elems/lane.
// ---------------------------------------------------------------------------
__global__ void layernorm_bf16(const float* __restrict__ h, const float* __restrict__ g,
                               const float* __restrict__ bta, bf16_t* __restrict__ out, int rows)
{
  int row  = blockIdx.x * (blockDim.x >> 5) + (threadIdx.x >> 5);
  int lane = threadIdx.x & 31;
  if (row >= rows) return;
  const float* p = h + (long)row * 64;
  float x0 = p[lane], x1 = p[lane + 32];
  float s = x0 + x1;
  for (int o = 16; o; o >>= 1) s += __shfl_xor(s, o, 32);
  float mu = s * (1.f / 64.f);
  float d0 = x0 - mu, d1 = x1 - mu;
  float v = d0 * d0 + d1 * d1;
  for (int o = 16; o; o >>= 1) v += __shfl_xor(v, o, 32);
  float rstd = rsqrtf(v * (1.f / 64.f) + 1e-5f);
  bf16_t* op = out + (long)row * 64;
  op[lane]      = f2bf(d0 * rstd * g[lane]      + bta[lane]);
  op[lane + 32] = f2bf(d1 * rstd * g[lane + 32] + bta[lane + 32]);
}

// ---------------------------------------------------------------------------
// Depthwise causal conv1d (k=4, left pad 3) + silu; writes f32 and bf16 copies.
// inproj: (L x 256), xi = cols [0,128).
// ---------------------------------------------------------------------------
__global__ void dwconv_silu(const float* __restrict__ inproj, const float* __restrict__ w,
                            const float* __restrict__ bias, float* __restrict__ xs,
                            bf16_t* __restrict__ xsb, int L)
{
  int idx = blockIdx.x * blockDim.x + threadIdx.x;
  if (idx >= L * 128) return;
  int d = idx & 127, l = idx >> 7;
  float acc = bias[d];
#pragma unroll
  for (int j = 0; j < 4; ++j) {
    int ls = l - 3 + j;
    if (ls >= 0) acc += w[d * 4 + j] * inproj[(long)ls * 256 + d];
  }
  float sv = acc / (1.f + expf(-acc));
  xs[idx]  = sv;
  xsb[idx] = f2bf(sv);
}

__global__ void softplus_k(float* __restrict__ t, int n)
{
  int i = blockIdx.x * blockDim.x + threadIdx.x;
  if (i >= n) return;
  float x = t[i];
  t[i] = (x > 20.f) ? x : log1pf(expf(x));
}

// ---------------------------------------------------------------------------
// Selective scan: serial over L, parallel over 128 channels (1 block).
// dbl: (L x 48): cols [4,20)=B, [20,36)=C. inproj: (L x 256), z = cols [128,256).
// Writes gated output directly as bf16 (A operand of out_proj).
// ---------------------------------------------------------------------------
__global__ void selective_scan(const float* __restrict__ delta, const float* __restrict__ xs,
                               const float* __restrict__ dbl, const float* __restrict__ A_log,
                               const float* __restrict__ Dp, const float* __restrict__ inproj,
                               bf16_t* __restrict__ ybf, int L)
{
  int d = threadIdx.x;   // 128 threads
  float A[16], hs[16];
#pragma unroll
  for (int s = 0; s < 16; ++s) { A[s] = -expf(A_log[d * 16 + s]); hs[s] = 0.f; }
  float Dd = Dp[d];
  for (int l = 0; l < L; ++l) {
    float dt = delta[(long)l * 128 + d];
    float xv = xs[(long)l * 128 + d];
    float dx = dt * xv;
    const float* bl = dbl + (long)l * 48;
    float y = 0.f;
#pragma unroll
    for (int s = 0; s < 16; ++s) {
      float hb = expf(dt * A[s]) * hs[s] + dx * bl[4 + s];
      hs[s] = hb;
      y += hb * bl[20 + s];
    }
    y += xv * Dd;
    float z = inproj[(long)l * 256 + 128 + d];
    y *= z / (1.f + expf(-z));
    ybf[(long)l * 128 + d] = f2bf(y);
  }
}

// ---------------------------------------------------------------------------
// Linear attention pieces.
// q: softmax over 64-dim head feature axis, scaled by 1/8, -> bf16.
// One wave per (row, head): L*8 waves total.
// ---------------------------------------------------------------------------
__global__ void q_softmax(const float* __restrict__ q, bf16_t* __restrict__ out, int L)
{
  int w    = blockIdx.x * (blockDim.x >> 5) + (threadIdx.x >> 5);
  int lane = threadIdx.x & 31;
  if (w >= L * 8) return;
  int row = w >> 3, hd = w & 7;
  const float* p = q + (long)row * 512 + hd * 64;
  float x0 = p[lane], x1 = p[lane + 32];
  float mx = fmaxf(x0, x1);
  for (int o = 16; o; o >>= 1) mx = fmaxf(mx, __shfl_xor(mx, o, 32));
  float e0 = expf(x0 - mx), e1 = expf(x1 - mx);
  float s = e0 + e1;
  for (int o = 16; o; o >>= 1) s += __shfl_xor(s, o, 32);
  float inv = 0.125f / s;      // DIM_HEAD^-0.5 = 1/8 folded in
  bf16_t* op = out + (long)row * 512 + hd * 64;
  op[lane] = f2bf(e0 * inv); op[lane + 32] = f2bf(e1 * inv);
}

// k: softmax over the sequence axis (per column). Pass 1: max & sum per column.
__global__ void k_col_stats(const float* __restrict__ k, float* __restrict__ cmax,
                            float* __restrict__ csum, int L)
{
  __shared__ float red[256];
  int c = blockIdx.x, t = threadIdx.x;
  float mx = -3.4e38f;
  for (int l = t; l < L; l += 256) mx = fmaxf(mx, k[(long)l * 512 + c]);
  red[t] = mx; __syncthreads();
  for (int s = 128; s > 0; s >>= 1) { if (t < s) red[t] = fmaxf(red[t], red[t + s]); __syncthreads(); }
  float m = red[0]; __syncthreads();
  float sm = 0.f;
  for (int l = t; l < L; l += 256) sm += expf(k[(long)l * 512 + c] - m);
  red[t] = sm; __syncthreads();
  for (int s = 128; s > 0; s >>= 1) { if (t < s) red[t] += red[t + s]; __syncthreads(); }
  if (t == 0) { cmax[c] = m; csum[c] = red[0]; }
}

__global__ void k_col_apply(const float* __restrict__ k, const float* __restrict__ cmax,
                            const float* __restrict__ csum, bf16_t* __restrict__ out, int L)
{
  int idx = blockIdx.x * blockDim.x + threadIdx.x;
  if (idx >= L * 512) return;
  int c = idx & 511;
  out[idx] = f2bf(expf(k[idx] - cmax[c]) / csum[c]);
}

// ctx[h] (64x64) = sum_l k[l,hd]^T v[l,he]; written TRANSPOSED (Bt[n=e][k=d]) as bf16.
__global__ void ctx_kernel(const bf16_t* __restrict__ kbf, const bf16_t* __restrict__ vbf,
                           bf16_t* __restrict__ ctxb, int L)
{
  int hd = blockIdx.x;          // 8 heads
  int t  = threadIdx.x;         // 256 threads
  int i  = t & 63;              // d index (k feature)
  int j0 = (t >> 6) << 4;       // e block of 16 (v feature)
  int rr = t >> 6, cc = t & 63;
  __shared__ float kl[4][64], vl[4][64];
  float acc[16];
#pragma unroll
  for (int j = 0; j < 16; ++j) acc[j] = 0.f;
  for (int l0 = 0; l0 < L; l0 += 4) {
    __syncthreads();
    kl[rr][cc] = bf2f(kbf[(long)(l0 + rr) * 512 + hd * 64 + cc]);
    vl[rr][cc] = bf2f(vbf[(long)(l0 + rr) * 512 + hd * 64 + cc]);
    __syncthreads();
#pragma unroll
    for (int r = 0; r < 4; ++r) {
      float kv = kl[r][i];
#pragma unroll
      for (int j = 0; j < 16; ++j) acc[j] += kv * vl[r][j0 + j];
    }
  }
#pragma unroll
  for (int j = 0; j < 16; ++j)
    ctxb[(long)hd * 4096 + (long)(j0 + j) * 64 + i] = f2bf(acc[j]);
}

__global__ void vec_add(float* __restrict__ a, const float* __restrict__ b, int n)
{
  int i = blockIdx.x * blockDim.x + threadIdx.x;
  if (i < n) a[i] += b[i];
}

// ===========================================================================
extern "C" void kernel_launch(void* const* d_in, const int* in_sizes, int n_in,
                              void* d_out, int out_size, void* d_ws, size_t ws_size,
                              hipStream_t stream)
{
  (void)in_sizes; (void)n_in; (void)out_size; (void)ws_size;
  const int L = LTOK;

  // ---- input mapping (setup_inputs dict order, depth-first) ----
  const float* X = (const float*)d_in[0];
  const float* convW[3] = {(const float*)d_in[1], (const float*)d_in[3], (const float*)d_in[5]};
  const float* convB[3] = {(const float*)d_in[2], (const float*)d_in[4], (const float*)d_in[6]};
  const float* scW  = (const float*)d_in[7];   const float* scB  = (const float*)d_in[8];
  const float* embW = (const float*)d_in[9];   const float* embB = (const float*)d_in[10];
  const float* mlnG = (const float*)d_in[11];  const float* mlnB = (const float*)d_in[12];
  const float* inpW = (const float*)d_in[13];  const float* inpB = (const float*)d_in[14];
  const float* cw   = (const float*)d_in[15];  const float* cb   = (const float*)d_in[16];
  const float* xpW  = (const float*)d_in[17];  const float* xpB  = (const float*)d_in[18];
  const float* dtW  = (const float*)d_in[19];  const float* dtB  = (const float*)d_in[20];
  const float* Alog = (const float*)d_in[21];  const float* Dp   = (const float*)d_in[22];
  const float* opW  = (const float*)d_in[23];  const float* opB  = (const float*)d_in[24];
  const float* outW = (const float*)d_in[57];  const float* outB = (const float*)d_in[58];

  // ---- workspace carve ----
  uintptr_t cur = (uintptr_t)d_ws;
  auto alloc = [&](size_t bytes) -> void* {
    uintptr_t p = (cur + 255) & ~(uintptr_t)255;
    cur = p + bytes;
    return (void*)p;
  };
  float*  F_tok = (float*)alloc((size_t)L * 32 * 4);
  float*  F_h   = (float*)alloc((size_t)L * 64 * 4);
  float*  F_g   = (float*)alloc((size_t)L * 512 * 4);
  float*  F_xs  = (float*)alloc((size_t)L * 128 * 4);
  float*  F_dbl = (float*)alloc((size_t)L * 48 * 4);
  float*  F_dlt = (float*)alloc((size_t)L * 128 * 4);
  float*  F_tmp = (float*)alloc((size_t)L * 64 * 4);
  float*  F_cmax = (float*)alloc(512 * 4);
  float*  F_csum = (float*)alloc(512 * 4);
  bf16_t* A0 = (bf16_t*)alloc((size_t)L * 512 * 2);
  bf16_t* A1 = (bf16_t*)alloc((size_t)L * 512 * 2);
  bf16_t* A2 = (bf16_t*)alloc((size_t)L * 512 * 2);
  bf16_t* CtxB = (bf16_t*)alloc((size_t)8 * 64 * 64 * 2);

  auto prepw = [&](const float* w, int K, int N) -> bf16_t* {
    int Kp = (K + 31) & ~31, Np = (N + 15) & ~15;
    bf16_t* dst = (bf16_t*)alloc((size_t)Kp * Np * 2);
    int tot = Kp * Np;
    prep_weight<<<dim3((tot + 255) / 256), dim3(256), 0, stream>>>(w, dst, K, N, Kp, Np);
    return dst;
  };

  auto gemm = [&](const bf16_t* A, int lda, const bf16_t* Bt, int Kpad,
                  const float* bias, int nBias, float* C, int ldc, int Npad,
                  int batches, long sA, long sB, long sC) {
    if (Npad % 64 == 0) {
      int tiles = (L / 16) * (Npad / 64);
      gemm_bf16_wmma<4><<<dim3((tiles + 3) / 4, 1, batches), dim3(128), 0, stream>>>(
          A, lda, Bt, Kpad, bias, nBias, C, ldc, L, Npad, Kpad, sA, sB, sC);
    } else if (Npad % 32 == 0) {
      int tiles = (L / 16) * (Npad / 32);
      gemm_bf16_wmma<2><<<dim3((tiles + 3) / 4, 1, batches), dim3(128), 0, stream>>>(
          A, lda, Bt, Kpad, bias, nBias, C, ldc, L, Npad, Kpad, sA, sB, sC);
    } else {
      int tiles = (L / 16) * (Npad / 16);
      gemm_bf16_wmma<1><<<dim3((tiles + 3) / 4, 1, batches), dim3(128), 0, stream>>>(
          A, lda, Bt, Kpad, bias, nBias, C, ldc, L, Npad, Kpad, sA, sB, sC);
    }
  };

  // ---- weight prep (bf16, transposed, K padded to 32 / N padded to 16) ----
  bf16_t* Wemb = prepw(embW, 32, 64);
  bf16_t* Winp = prepw(inpW, 64, 256);
  bf16_t* Wxp  = prepw(xpW, 128, 36);
  bf16_t* Wdt  = prepw(dtW, 4, 128);
  bf16_t* Wop  = prepw(opW, 128, 64);
  bf16_t *Wq[2], *Wk[2], *Wv[2], *Wo[2], *Wf1[2], *Wf2[2];
  for (int li = 0; li < 2; ++li) {
    int base = 25 + 16 * li;
    Wq[li]  = prepw((const float*)d_in[base + 2],  64, 512);
    Wk[li]  = prepw((const float*)d_in[base + 4],  64, 512);
    Wv[li]  = prepw((const float*)d_in[base + 6],  64, 512);
    Wo[li]  = prepw((const float*)d_in[base + 8], 512,  64);
    Wf1[li] = prepw((const float*)d_in[base + 12], 64, 256);
    Wf2[li] = prepw((const float*)d_in[base + 14], 256, 64);
  }
  bf16_t* Wout = prepw(outW, 64, 32);

  // ---- shortcut conv initializes d_out ----
  conv3d_shortcut<<<(2 * 32 * L + 255) / 256, 256, 0, stream>>>(X, scW, scB, (float*)d_out);

  const int ks[3] = {1, 3, 5};
  for (int br = 0; br < 3; ++br)
  for (int b = 0; b < 2; ++b) {
    // conv branch -> tokens, leaky relu
    conv3d_branch<<<(32 * L + 255) / 256, 256, 0, stream>>>(X, convW[br], convB[br], F_tok, b, ks[br]);
    convert_slice<<<(L * 32 + 255) / 256, 256, 0, stream>>>(F_tok, 32, 0, 32, A0, 32, L, 0);
    gemm(A0, 32, Wemb, 32, embB, 64, F_h, 64, 64, 1, 0, 0, 0);

    // ---- Mamba block ----
    layernorm_bf16<<<(L + 7) / 8, 256, 0, stream>>>(F_h, mlnG, mlnB, A0, L);
    gemm(A0, 64, Winp, 64, inpB, 256, F_g, 256, 256, 1, 0, 0, 0);
    dwconv_silu<<<(L * 128 + 255) / 256, 256, 0, stream>>>(F_g, cw, cb, F_xs, A0, L);
    gemm(A0, 128, Wxp, 128, xpB, 36, F_dbl, 48, 48, 1, 0, 0, 0);
    convert_slice<<<(L * 32 + 255) / 256, 256, 0, stream>>>(F_dbl, 48, 0, 4, A1, 32, L, 0);
    gemm(A1, 32, Wdt, 32, dtB, 128, F_dlt, 128, 128, 1, 0, 0, 0);
    softplus_k<<<(L * 128 + 255) / 256, 256, 0, stream>>>(F_dlt, L * 128);
    selective_scan<<<1, 128, 0, stream>>>(F_dlt, F_xs, F_dbl, Alog, Dp, F_g, A0, L);
    gemm(A0, 128, Wop, 128, opB, 64, F_tmp, 64, 64, 1, 0, 0, 0);
    vec_add<<<(L * 64 + 255) / 256, 256, 0, stream>>>(F_h, F_tmp, L * 64);

    // ---- transformer layers: linear attention + FF ----
    for (int li = 0; li < 2; ++li) {
      int base = 25 + 16 * li;
      const float* laG = (const float*)d_in[base + 0];
      const float* laB = (const float*)d_in[base + 1];
      const float* qB  = (const float*)d_in[base + 3];
      const float* kB  = (const float*)d_in[base + 5];
      const float* vB  = (const float*)d_in[base + 7];
      const float* oB  = (const float*)d_in[base + 9];
      const float* lfG = (const float*)d_in[base + 10];
      const float* lfB = (const float*)d_in[base + 11];
      const float* f1B = (const float*)d_in[base + 13];
      const float* f2B = (const float*)d_in[base + 15];

      layernorm_bf16<<<(L + 7) / 8, 256, 0, stream>>>(F_h, laG, laB, A0, L);
      // k -> sequence softmax -> A1
      gemm(A0, 64, Wk[li], 64, kB, 512, F_g, 512, 512, 1, 0, 0, 0);
      k_col_stats<<<512, 256, 0, stream>>>(F_g, F_cmax, F_csum, L);
      k_col_apply<<<(L * 512 + 255) / 256, 256, 0, stream>>>(F_g, F_cmax, F_csum, A1, L);
      // v -> A2
      gemm(A0, 64, Wv[li], 64, vB, 512, F_g, 512, 512, 1, 0, 0, 0);
      convert_slice<<<(L * 512 + 255) / 256, 256, 0, stream>>>(F_g, 512, 0, 512, A2, 512, L, 0);
      // q -> feature softmax -> A0 (ln output fully consumed)
      gemm(A0, 64, Wq[li], 64, qB, 512, F_g, 512, 512, 1, 0, 0, 0);
      q_softmax<<<L, 256, 0, stream>>>(F_g, A0, L);   // L*8 waves, 8 waves/block
      // ctx = k^T v per head; o = q @ ctx (batched per-head WMMA GEMM)
      ctx_kernel<<<8, 256, 0, stream>>>(A1, A2, CtxB, L);
      gemm(A0, 512, CtxB, 64, (const float*)nullptr, 0, F_g, 512, 64, 8, 64, 4096, 64);
      convert_slice<<<(L * 512 + 255) / 256, 256, 0, stream>>>(F_g, 512, 0, 512, A1, 512, L, 0);
      gemm(A1, 512, Wo[li], 512, oB, 64, F_tmp, 64, 64, 1, 0, 0, 0);
      vec_add<<<(L * 64 + 255) / 256, 256, 0, stream>>>(F_h, F_tmp, L * 64);
      // FF
      layernorm_bf16<<<(L + 7) / 8, 256, 0, stream>>>(F_h, lfG, lfB, A0, L);
      gemm(A0, 64, Wf1[li], 64, f1B, 256, F_g, 256, 256, 1, 0, 0, 0);
      convert_slice<<<(L * 256 + 255) / 256, 256, 0, stream>>>(F_g, 256, 0, 256, A0, 256, L, 1);
      gemm(A0, 256, Wf2[li], 256, f2B, 64, F_tmp, 64, 64, 1, 0, 0, 0);
      vec_add<<<(L * 64 + 255) / 256, 256, 0, stream>>>(F_h, F_tmp, L * 64);
    }

    // out projection + accumulate into d_out for this batch
    convert_slice<<<(L * 64 + 255) / 256, 256, 0, stream>>>(F_h, 64, 0, 64, A0, 64, L, 0);
    gemm(A0, 64, Wout, 64, outB, 32, F_g, 32, 32, 1, 0, 0, 0);
    accum_out<<<(32 * L + 255) / 256, 256, 0, stream>>>(F_g, (float*)d_out + (long)b * 32 * L, L);
  }
}
